// MultiHeadAttention_85461259256560
// MI455X (gfx1250) — matrix operand order repaired and checked by reading, not compiled, not run on hardware
//
#include <hip/hip_runtime.h>

// MI455X / gfx1250 multi-head attention forward.
// Matmuls via v_wmma_f32_16x16x32_f16 (wave32 WMMA); tile staging via the
// Tensor Data Mover (tensor_load_to_lds + s_wait_tensorcnt) where the tile is
// a plain strided 2D f16 copy, plus global_prefetch hints.
// Pipeline: qkv_kernel (x3) -> flash attn_kernel -> proj_kernel.

typedef __attribute__((ext_vector_type(16))) _Float16 v16h;
typedef __attribute__((ext_vector_type(8)))  _Float16 v8h;
typedef __attribute__((ext_vector_type(8)))  float    v8f;
typedef __attribute__((ext_vector_type(4)))  unsigned int u32x4;
typedef __attribute__((ext_vector_type(8)))  int          i32x8;
typedef __attribute__((ext_vector_type(4)))  int          i32x4;

constexpr int Bn = 4, Sn = 2048, En = 1024, Hn = 16, Dn = 64;

__device__ __forceinline__ v8f wmma16(v16h a, v16h b, v8f c) {
  // (neg_a, A, neg_b, B, c_mod, C, reuse_a, reuse_b)
  return __builtin_amdgcn_wmma_f32_16x16x32_f16(false, a, false, b, (short)0, c,
                                                false, false);
}

__device__ __forceinline__ v8f zero8() {
  v8f z;
#pragma unroll
  for (int i = 0; i < 8; ++i) z[i] = 0.0f;
  return z;
}

// ---------------------------------------------------------------------------
// Tensor Data Mover: DMA one f16 2D tile global->LDS.
//   width        : tile width in f16 elements (row length, mult of 2 dwords)
//   rows         : tile height
//   stride       : global row stride in f16 elements
//   pi, pa       : D# pad_interval / pad_amount codes (ISA 8.4) giving the
//                  padded LDS row stride used by the fragment loaders.
// Wave-uniform args; issue from a single wave, then s_wait_tensorcnt 0.
// ---------------------------------------------------------------------------
__device__ __forceinline__ void tdm_load_tile_f16(void* lds_dst, const void* gsrc,
                                                  unsigned width, unsigned rows,
                                                  unsigned stride, unsigned pi,
                                                  unsigned pa) {
  const unsigned long long ga = (unsigned long long)(uintptr_t)gsrc;
  u32x4 g0;
  g0[0] = 1u;                                    // count=1, user descriptor
  g0[1] = (unsigned)(uintptr_t)lds_dst;          // LDS byte offset (addr[31:0])
  g0[2] = (unsigned)ga;                          // global_addr[31:0]
  g0[3] = (unsigned)((ga >> 32) & 0x01ffffffu) | (2u << 30);  // type=2 (image)
  i32x8 g1;
  g1[0] = (int)((1u << 16)      // data_size = 1 -> 2 bytes
              | (1u << 20)      // pad_enable
              | (pi << 22)      // pad_interval
              | (pa << 25));    // pad_amount
  g1[1] = (int)(width << 16);   // [47:32]=bar addr 0 | [63:48]=tensor_dim0 lo
  g1[2] = (int)(rows << 16);    // [79:64]=tensor_dim0 hi 0 | [95:80]=tensor_dim1 lo
  g1[3] = (int)(width << 16);   // [111:96]=tensor_dim1 hi 0 | [127:112]=tile_dim0
  g1[4] = (int)rows;            // [143:128]=tile_dim1 | [159:144]=tile_dim2=0
  g1[5] = (int)stride;          // tensor_dim0_stride[31:0]
  g1[6] = 0;
  g1[7] = 0;
  i32x4 gz;
  gz[0] = gz[1] = gz[2] = gz[3] = 0;
#if defined(__clang_major__) && __clang_major__ >= 23
  i32x8 gz8;
#pragma unroll
  for (int i = 0; i < 8; ++i) gz8[i] = 0;
  __builtin_amdgcn_tensor_load_to_lds(g0, g1, gz, gz, gz8, 0);
#else
  __builtin_amdgcn_tensor_load_to_lds(g0, g1, gz, gz, 0);
#endif
}

// A-matrix fragment 16x32 f16 (row-major source, stride in halves).
// Lane l holds row (l&15); K groups: [(l>>4)*8 .. +7] and [+16 .. +23].
__device__ __forceinline__ v16h ld_fragA(const _Float16* base, int stride, int k0) {
  const int lane = threadIdx.x & 31;
  const _Float16* p = base + (lane & 15) * stride + k0 + ((lane >> 4) << 3);
  v8h lo = *(const v8h*)(p);
  v8h hi = *(const v8h*)(p + 16);
  v16h r;
#pragma unroll
  for (int i = 0; i < 8; ++i) { r[i] = lo[i]; r[i + 8] = hi[i]; }
  return r;
}

// B-matrix fragment 32x16 f16 from [n][k] (column-of-B major) storage.
// Lane l holds column n=(l&15); K groups: [(l>>4)*8 .. +7] and [+16 .. +23].
__device__ __forceinline__ v16h ld_fragB(const _Float16* base, int stride, int k0) {
  const int lane = threadIdx.x & 31;
  const _Float16* p = base + (lane & 15) * stride + k0 + ((lane >> 4) << 3);
  v8h lo = *(const v8h*)(p);
  v8h hi = *(const v8h*)(p + 16);
  v16h r;
#pragma unroll
  for (int i = 0; i < 8; ++i) { r[i] = lo[i]; r[i + 8] = hi[i]; }
  return r;
}

// Reductions across the 16-lane halves of a wave32 (C-layout rows).
__device__ __forceinline__ float rmax16(float x) {
  x = fmaxf(x, __shfl_xor(x, 1));
  x = fmaxf(x, __shfl_xor(x, 2));
  x = fmaxf(x, __shfl_xor(x, 4));
  x = fmaxf(x, __shfl_xor(x, 8));
  return x;
}
__device__ __forceinline__ float rsum16(float x) {
  x += __shfl_xor(x, 1);
  x += __shfl_xor(x, 2);
  x += __shfl_xor(x, 4);
  x += __shfl_xor(x, 8);
  return x;
}

// ---------------------------------------------------------------------------
// Kernel 1: per-head projection  out[b,h,s,d] = (x[b,s,:] @ W[h,:,d] + bias)*scale
// grid = (Sn/128, Bn*Hn), block = 256 (8 waves). scale folds 1/sqrt(D) into Q.
// ---------------------------------------------------------------------------
__global__ __launch_bounds__(256) void qkv_kernel(const float* __restrict__ x,
                                                  const float* __restrict__ W,
                                                  const float* __restrict__ bias,
                                                  _Float16* __restrict__ out,
                                                  float scale) {
  __shared__ alignas(16) _Float16 ldsA[128 * 40];  // x tile  [128 rows][32 k]
  __shared__ alignas(16) _Float16 ldsB[64 * 40];   // W^T     [64 d][32 e]
  const int tid = threadIdx.x;
  const int lane = tid & 31, wave = tid >> 5;
  const int m0 = blockIdx.x * 128;
  const int b = blockIdx.y / Hn, h = blockIdx.y % Hn;
  const float* xb = x + (size_t)b * Sn * En;
  const float* Wh = W + (size_t)h * En * Dn;

  v8f acc[4];
#pragma unroll
  for (int j = 0; j < 4; ++j) acc[j] = zero8();

  for (int e0 = 0; e0 < En; e0 += 32) {
    if (e0 + 32 < En) {  // prefetch next x chunk (f32): 128 rows x 32 cols
      const float* np = xb + (size_t)(m0 + (tid >> 1)) * En + e0 + 32 + (tid & 1) * 16;
      __builtin_prefetch(np, 0, 0);
    }
    __syncthreads();
    {  // stage x tile: f32 -> f16
      const int row = tid >> 1;
      const int k0 = (tid & 1) * 16;
      const float* src = xb + (size_t)(m0 + row) * En + e0 + k0;
      _Float16* dst = ldsA + row * 40 + k0;
#pragma unroll
      for (int j = 0; j < 16; ++j) dst[j] = (_Float16)src[j];
    }
    {  // stage W chunk transposed: ldsB[d][e] = W[e0+e][d]
      const int e = tid >> 3;
      const int d0 = (tid & 7) * 8;
      const float* src = Wh + (size_t)(e0 + e) * Dn + d0;
#pragma unroll
      for (int j = 0; j < 8; ++j) ldsB[(d0 + j) * 40 + e] = (_Float16)src[j];
    }
    __syncthreads();
    v16h a = ld_fragA(ldsA + wave * 16 * 40, 40, 0);
#pragma unroll
    for (int j = 0; j < 4; ++j) {
      v16h bf = ld_fragB(ldsB + j * 16 * 40, 40, 0);
      acc[j] = wmma16(a, bf, acc[j]);
    }
  }

  // epilogue: bias + scale, f16 store to [b,h,s,d]
  const int colb = lane & 15;
  const int rtop = (lane >> 4) * 8;
  const float* bh = bias + h * Dn;
  _Float16* op = out + (((size_t)b * Hn + h) * Sn + m0 + wave * 16) * Dn;
#pragma unroll
  for (int j = 0; j < 4; ++j) {
    const float bv = bh[j * 16 + colb];
#pragma unroll
    for (int r = 0; r < 8; ++r) {
      op[(size_t)(rtop + r) * Dn + j * 16 + colb] =
          (_Float16)((acc[j][r] + bv) * scale);
    }
  }
}

// ---------------------------------------------------------------------------
// Kernel 2: flash attention. grid = (Sn/128, Bn*Hn), block = 256 (8 waves).
// Each wave owns 16 query rows; loops keys in 32-column chunks with online
// softmax kept entirely in the WMMA C-register layout (row == VGPR index).
// K tile staged by the TDM; V tile transposed by lanes; next chunk prefetched.
// ---------------------------------------------------------------------------
__global__ __launch_bounds__(256) void attn_kernel(const _Float16* __restrict__ Q,
                                                   const _Float16* __restrict__ K,
                                                   const _Float16* __restrict__ V,
                                                   _Float16* __restrict__ O) {
  __shared__ alignas(16) _Float16 ldsK[32 * 72];       // K rows   [32 t][64 d] (TDM pad)
  __shared__ alignas(16) _Float16 ldsV[64 * 40];       // V^T      [64 d][32 t]
  __shared__ alignas(16) _Float16 ldsP[8 * 16 * 40];   // per-wave P [16][32]
  const int tid = threadIdx.x;
  const int lane = tid & 31, wave = tid >> 5;
  const int bh = blockIdx.y;
  const int b = bh / Hn, h = bh % Hn;
  const int q0 = blockIdx.x * 128 + wave * 16;

  const _Float16* Qp = Q + ((size_t)bh * Sn + q0) * Dn;
  const _Float16* Kp = K + (size_t)bh * Sn * Dn;
  const _Float16* Vp = V + (size_t)bh * Sn * Dn;

  // Q fragments stay in registers (scale pre-folded by qkv_kernel).
  const v16h aq0 = ld_fragA(Qp, Dn, 0);
  const v16h aq1 = ld_fragA(Qp, Dn, 32);

  v8f acc[4];
#pragma unroll
  for (int j = 0; j < 4; ++j) acc[j] = zero8();
  float m_r[8], l_r[8];
#pragma unroll
  for (int r = 0; r < 8; ++r) { m_r[r] = -3.0e30f; l_r[r] = 0.0f; }

  _Float16* Pw = ldsP + wave * 16 * 40;

  for (int t0 = 0; t0 < Sn; t0 += 32) {
    __syncthreads();  // previous chunk's LDS consumers are done
    if (wave == 0) {  // TDM: K chunk [32 t][64 d] -> ldsK (stride 72 via pad)
      tdm_load_tile_f16(ldsK, Kp + (size_t)t0 * Dn,
                        /*width=*/64, /*rows=*/32, /*stride=*/Dn,
                        /*pad_interval: 32 dw*/ 4, /*pad_amount: 4 dw*/ 3);
    }
    if (t0 + 32 < Sn) {  // prefetch next K/V chunk (4 KB each)
      __builtin_prefetch(Kp + (size_t)(t0 + 32) * Dn + (size_t)tid * 8, 0, 0);
      __builtin_prefetch(Vp + (size_t)(t0 + 32) * Dn + (size_t)tid * 8, 0, 0);
    }
    {  // V chunk transposed by lanes: ldsV[d][t]
      const int t = tid >> 3, d0 = (tid & 7) * 8;
      const v8h vv = *(const v8h*)(Vp + (size_t)(t0 + t) * Dn + d0);
#pragma unroll
      for (int j = 0; j < 8; ++j) ldsV[(d0 + j) * 40 + t] = vv[j];
    }
    if (wave == 0) __builtin_amdgcn_s_wait_tensorcnt(0);
    __syncthreads();

    // scores S = Q K^T : two 16x16 tiles over D=64 (two K-steps each)
    v8f s0 = zero8(), s1 = zero8();
    s0 = wmma16(aq0, ld_fragB(ldsK, 72, 0), s0);
    s0 = wmma16(aq1, ld_fragB(ldsK, 72, 32), s0);
    s1 = wmma16(aq0, ld_fragB(ldsK + 16 * 72, 72, 0), s1);
    s1 = wmma16(aq1, ld_fragB(ldsK + 16 * 72, 72, 32), s1);

    // online softmax in C layout (VGPR r <-> row r + 8*(lane>>4))
    float alpha[8];
#pragma unroll
    for (int r = 0; r < 8; ++r) {
      const float cm = rmax16(fmaxf(s0[r], s1[r]));
      const float mn = fmaxf(m_r[r], cm);
      alpha[r] = __expf(m_r[r] - mn);
      m_r[r] = mn;
      s0[r] = __expf(s0[r] - mn);
      s1[r] = __expf(s1[r] - mn);
      const float rs = rsum16(s0[r] + s1[r]);
      l_r[r] = l_r[r] * alpha[r] + rs;
    }
#pragma unroll
    for (int j = 0; j < 4; ++j)
#pragma unroll
      for (int r = 0; r < 8; ++r) acc[j][r] *= alpha[r];

    // spill P (f16) to per-wave LDS to re-fragment as WMMA A operand
    const int colb = lane & 15;
    const int rtop = (lane >> 4) * 8;
#pragma unroll
    for (int r = 0; r < 8; ++r) {
      Pw[(rtop + r) * 40 + colb] = (_Float16)s0[r];
      Pw[(rtop + r) * 40 + 16 + colb] = (_Float16)s1[r];
    }
    __syncthreads();

    // O += P V
    const v16h ap = ld_fragA(Pw, 40, 0);
#pragma unroll
    for (int j = 0; j < 4; ++j) {
      v16h bv = ld_fragB(ldsV + j * 16 * 40, 40, 0);
      acc[j] = wmma16(ap, bv, acc[j]);
    }
  }

  // normalize + store concatenated heads: O[b, s, h*64 + d] (f16)
  const int colb = lane & 15;
  const int rtop = (lane >> 4) * 8;
  float inv[8];
#pragma unroll
  for (int r = 0; r < 8; ++r) inv[r] = 1.0f / l_r[r];
  _Float16* op = O + ((size_t)b * Sn + q0) * En + h * Dn;
#pragma unroll
  for (int j = 0; j < 4; ++j)
#pragma unroll
    for (int r = 0; r < 8; ++r)
      op[(size_t)(rtop + r) * En + j * 16 + colb] = (_Float16)(acc[j][r] * inv[r]);
}

// ---------------------------------------------------------------------------
// Kernel 3: output projection  out = concat @ Wf + bf  (8192 x 1024 x 1024)
// grid = (Bn*Sn/128, En/64), block = 256. A tile staged by the TDM.
// ---------------------------------------------------------------------------
__global__ __launch_bounds__(256) void proj_kernel(const _Float16* __restrict__ C,
                                                   const float* __restrict__ Wf,
                                                   const float* __restrict__ bf,
                                                   float* __restrict__ out) {
  __shared__ alignas(16) _Float16 ldsA[128 * 40];
  __shared__ alignas(16) _Float16 ldsB[64 * 40];
  const int tid = threadIdx.x;
  const int lane = tid & 31, wave = tid >> 5;
  const int m0 = blockIdx.x * 128;
  const int n0 = blockIdx.y * 64;

  v8f acc[4];
#pragma unroll
  for (int j = 0; j < 4; ++j) acc[j] = zero8();

  for (int e0 = 0; e0 < En; e0 += 32) {
    __syncthreads();
    if (wave == 0) {  // TDM: A tile [128 rows][32 k] f16 -> ldsA (stride 40 via pad)
      tdm_load_tile_f16(ldsA, C + (size_t)m0 * En + e0,
                        /*width=*/32, /*rows=*/128, /*stride=*/En,
                        /*pad_interval: 16 dw*/ 3, /*pad_amount: 4 dw*/ 3);
    }
    if (e0 + 32 < En) {  // prefetch next Wf chunk
      __builtin_prefetch(Wf + (size_t)(e0 + 32 + (tid >> 3)) * En + n0 + (tid & 7) * 8,
                         0, 0);
    }
    {  // B tile: Wf[e][n0+d] f32 -> ldsB[d][e] f16
      const int e = tid >> 3;
      const int d0 = (tid & 7) * 8;
      const float* src = Wf + (size_t)(e0 + e) * En + n0 + d0;
#pragma unroll
      for (int j = 0; j < 8; ++j) ldsB[(d0 + j) * 40 + e] = (_Float16)src[j];
    }
    if (wave == 0) __builtin_amdgcn_s_wait_tensorcnt(0);
    __syncthreads();
    v16h a = ld_fragA(ldsA + wave * 16 * 40, 40, 0);
#pragma unroll
    for (int j = 0; j < 4; ++j) {
      v16h bfr = ld_fragB(ldsB + j * 16 * 40, 40, 0);
      acc[j] = wmma16(a, bfr, acc[j]);
    }
  }

  const int colb = lane & 15;
  const int rtop = (lane >> 4) * 8;
  float* op = out + (size_t)(m0 + wave * 16) * En + n0;
#pragma unroll
  for (int j = 0; j < 4; ++j) {
    const float bb = bf[n0 + j * 16 + colb];
#pragma unroll
    for (int r = 0; r < 8; ++r)
      op[(size_t)(rtop + r) * En + j * 16 + colb] = acc[j][r] + bb;
  }
}

// ---------------------------------------------------------------------------
extern "C" void kernel_launch(void* const* d_in, const int* in_sizes, int n_in,
                              void* d_out, int out_size, void* d_ws, size_t ws_size,
                              hipStream_t stream) {
  (void)in_sizes; (void)n_in; (void)out_size; (void)ws_size;
  const float* x  = (const float*)d_in[0];
  const float* Wq = (const float*)d_in[1];
  const float* bq = (const float*)d_in[2];
  const float* Wk = (const float*)d_in[3];
  const float* bk = (const float*)d_in[4];
  const float* Wv = (const float*)d_in[5];
  const float* bv = (const float*)d_in[6];
  const float* Wf = (const float*)d_in[7];
  const float* bf = (const float*)d_in[8];
  float* out = (float*)d_out;

  const size_t NH = (size_t)Bn * Hn * Sn * Dn;  // == Bn*Sn*En
  _Float16* Qh = (_Float16*)d_ws;
  _Float16* Kh = Qh + NH;
  _Float16* Vh = Kh + NH;
  _Float16* Ch = Vh + NH;  // total 4*NH*2 = 64 MB of workspace

  const dim3 blk(256);
  const dim3 gproj(Sn / 128, Bn * Hn);
  const float qscale = 0.125f;  // 1/sqrt(D)

  qkv_kernel<<<gproj, blk, 0, stream>>>(x, Wq, bq, Qh, qscale);
  qkv_kernel<<<gproj, blk, 0, stream>>>(x, Wk, bk, Kh, 1.0f);
  qkv_kernel<<<gproj, blk, 0, stream>>>(x, Wv, bv, Vh, 1.0f);
  attn_kernel<<<gproj, blk, 0, stream>>>(Qh, Kh, Vh, Ch);
  proj_kernel<<<dim3((Bn * Sn) / 128, En / 64), blk, 0, stream>>>(Ch, Wf, bf, out);
}
